// RAFT_37976100831365
// MI455X (gfx1250) — matrix operand order repaired and checked by reading, not compiled, not run on hardware
//
#include <hip/hip_runtime.h>
#include <math.h>

// ---------------------------------------------------------------------------
// CDNA5 (gfx1250) RAFT forward. All convs / the correlation volume run as
// implicit GEMM on v_wmma_f32_16x16x32_bf16 (wave32, one 16x16 tile per wave,
// 4 waves per 128-thread block). f32 accumulate, bf16 operands.
// ---------------------------------------------------------------------------

typedef __attribute__((ext_vector_type(16))) __bf16 v16bf;
typedef __attribute__((ext_vector_type(8)))  float  v8f;

// ---------------------------------------------------------------------------
// Implicit-GEMM convolution (NCHW, OIHW), fused epilogue:
//   v = (acc + bias[n]) * out_scale; if(scale) v = v*scale[n]+shift[n];
//   if(resid) v += resid; if(relu) v = max(v,0)
// ---------------------------------------------------------------------------
__global__ void conv_wmma_kernel(
    const float* __restrict__ x, const float* __restrict__ wgt,
    const float* __restrict__ bias,
    const float* __restrict__ scale, const float* __restrict__ shift,
    const float* __restrict__ resid,
    float* __restrict__ y,
    int C, int H, int W, int O, int KH, int KW, int stride,
    int Hout, int Wout, int relu_flag, float out_scale)
{
  const int lane = threadIdx.x & 31;
  const int wave = threadIdx.x >> 5;
  const int half = lane >> 4;
  const int l16  = lane & 15;
  const int M    = Hout * Wout;
  const int KHW  = KH * KW;
  const int K    = C * KHW;
  const int m0   = blockIdx.x * 16;
  const int n0   = (blockIdx.y * 4 + wave) * 16;

  const int  am     = m0 + l16;
  const bool mvalid = am < M;
  const int  oh     = mvalid ? am / Wout : 0;
  const int  ow     = mvalid ? am % Wout : 0;
  const int  padh   = KH >> 1;
  const int  padw   = KW >> 1;
  const int  nn     = n0 + l16;
  const bool nvalid = nn < O;

  v8f acc = {0.f, 0.f, 0.f, 0.f, 0.f, 0.f, 0.f, 0.f};

  for (int k0 = 0; k0 < K; k0 += 32) {
    v16bf a, b;
#pragma unroll
    for (int e = 0; e < 16; ++e) {
      // A fragment: lane row = m, K per ISA 16-bit A 16x32 layout
      int ka = k0 + half * 8 + e + ((e < 8) ? 0 : 8);
      float av = 0.f;
      if (mvalid && (ka < K)) {
        int c  = ka / KHW;
        int r2 = ka - c * KHW;
        int kh = r2 / KW;
        int kw = r2 - kh * KW;
        int ih = oh * stride - padh + kh;
        int iw = ow * stride - padw + kw;
        if ((ih >= 0) && (ih < H) && (iw >= 0) && (iw < W))
          av = x[(size_t)c * H * W + (size_t)ih * W + iw];
      }
      a[e] = (__bf16)av;
      // B fragment: lane col = n, K per same 16x32 layout as A
      int kb = k0 + half * 8 + e + ((e < 8) ? 0 : 8);
      float bv = 0.f;
      if (nvalid && (kb < K))
        bv = wgt[(size_t)nn * K + kb];
      b[e] = (__bf16)bv;
    }
    acc = __builtin_amdgcn_wmma_f32_16x16x32_bf16(false, a, false, b,
                                                  (short)0, acc, false, false);
  }

#pragma unroll
  for (int r = 0; r < 8; ++r) {
    int m = m0 + r + half * 8;
    int n = n0 + l16;
    if ((m < M) && (n < O)) {
      float v = (acc[r] + bias[n]) * out_scale;
      if (scale) v = v * scale[n] + shift[n];
      if (resid) v += resid[(size_t)n * M + m];
      if (relu_flag) v = fmaxf(v, 0.f);
      y[(size_t)n * M + m] = v;
    }
  }
}

// ---------------------------------------------------------------------------
// TN GEMM for the correlation volume: C[m,n] = scl * sum_k A[k*M+m]*B[k*N+n]
// ---------------------------------------------------------------------------
__global__ void gemm_tn_wmma_kernel(const float* __restrict__ A,
                                    const float* __restrict__ B,
                                    float* __restrict__ Cc,
                                    int M, int N, int K, float scl)
{
  const int lane = threadIdx.x & 31;
  const int wave = threadIdx.x >> 5;
  const int half = lane >> 4;
  const int l16  = lane & 15;
  const int m0   = blockIdx.x * 16;
  const int n0   = (blockIdx.y * 4 + wave) * 16;
  const int am   = m0 + l16;
  const int bn   = n0 + l16;
  const bool mvalid = am < M;
  const bool nvalid = bn < N;

  v8f acc = {0.f, 0.f, 0.f, 0.f, 0.f, 0.f, 0.f, 0.f};
  for (int k0 = 0; k0 < K; k0 += 32) {
    v16bf a, b;
#pragma unroll
    for (int e = 0; e < 16; ++e) {
      int ka = k0 + half * 8 + e + ((e < 8) ? 0 : 8);
      float av = (mvalid && ka < K) ? A[(size_t)ka * M + am] : 0.f;
      a[e] = (__bf16)av;
      int kb = k0 + half * 8 + e + ((e < 8) ? 0 : 8);
      float bv = (nvalid && kb < K) ? B[(size_t)kb * N + bn] : 0.f;
      b[e] = (__bf16)bv;
    }
    acc = __builtin_amdgcn_wmma_f32_16x16x32_bf16(false, a, false, b,
                                                  (short)0, acc, false, false);
  }
#pragma unroll
  for (int r = 0; r < 8; ++r) {
    int m = m0 + r + half * 8;
    int n = n0 + l16;
    if ((m < M) && (n < N))
      Cc[(size_t)m * N + n] = acc[r] * scl;
  }
}

// ---------------------------------------------------------------------------
// InstanceNorm (affine=False) + optional residual + optional relu, in-place ok
// one block (256 threads) per channel
// ---------------------------------------------------------------------------
__global__ void instnorm_kernel(const float* __restrict__ x, float* __restrict__ y,
                                int HW, int relu_flag, const float* __restrict__ resid)
{
  const int c = blockIdx.x;
  const float* xc = x + (size_t)c * HW;
  __shared__ float ssum[256];
  __shared__ float ssq[256];
  float s = 0.f, q = 0.f;
  for (int i = threadIdx.x; i < HW; i += 256) {
    float v = xc[i];
    s += v; q += v * v;
  }
  ssum[threadIdx.x] = s; ssq[threadIdx.x] = q;
  __syncthreads();
  for (int off = 128; off > 0; off >>= 1) {
    if (threadIdx.x < (unsigned)off) {
      ssum[threadIdx.x] += ssum[threadIdx.x + off];
      ssq[threadIdx.x]  += ssq[threadIdx.x + off];
    }
    __syncthreads();
  }
  const float mu  = ssum[0] / (float)HW;
  const float var = ssq[0] / (float)HW - mu * mu;
  const float inv = rsqrtf(var + 1e-5f);
  float* yc = y + (size_t)c * HW;
  const float* rc = resid ? (resid + (size_t)c * HW) : nullptr;
  for (int i = threadIdx.x; i < HW; i += 256) {
    float v = (xc[i] - mu) * inv;
    if (rc) v += rc[i];
    if (relu_flag) v = fmaxf(v, 0.f);
    yc[i] = v;
  }
}

// ---------------------------------------------------------------------------
// Elementwise kernels
// ---------------------------------------------------------------------------
__global__ void ew_copy_kernel(float* dst, const float* src, int n) {
  int i = blockIdx.x * blockDim.x + threadIdx.x;
  if (i < n) dst[i] = src[i];
}
__global__ void ew_add_kernel(float* a, const float* b, int n) {
  int i = blockIdx.x * blockDim.x + threadIdx.x;
  if (i < n) a[i] += b[i];
}
__global__ void ew_tanh_kernel(float* dst, const float* src, int n) {
  int i = blockIdx.x * blockDim.x + threadIdx.x;
  if (i < n) dst[i] = tanhf(src[i]);
}
__global__ void ew_relu_kernel(float* dst, const float* src, int n) {
  int i = blockIdx.x * blockDim.x + threadIdx.x;
  if (i < n) dst[i] = fmaxf(src[i], 0.f);
}
__global__ void ew_sigmul_kernel(float* dst, const float* a, const float* b, int n) {
  int i = blockIdx.x * blockDim.x + threadIdx.x;
  if (i < n) dst[i] = (1.f / (1.f + expf(-a[i]))) * b[i];
}
__global__ void ew_gru_kernel(float* net, const float* zc, const float* qc, int n) {
  int i = blockIdx.x * blockDim.x + threadIdx.x;
  if (i < n) {
    float z = 1.f / (1.f + expf(-zc[i]));
    net[i] = (1.f - z) * net[i] + z * tanhf(qc[i]);
  }
}
__global__ void coords_init_kernel(float* c, int h, int w) {
  int i = blockIdx.x * blockDim.x + threadIdx.x;
  int HW = h * w;
  if (i < HW) {
    c[i]      = (float)(i % w);   // x
    c[HW + i] = (float)(i / w);   // y
  }
}
__global__ void flow_from_coords_kernel(const float* c, float* f, int h, int w) {
  int i = blockIdx.x * blockDim.x + threadIdx.x;
  int HW = h * w;
  if (i < HW) {
    f[i]      = c[i]      - (float)(i % w);
    f[HW + i] = c[HW + i] - (float)(i / w);
  }
}

// ---------------------------------------------------------------------------
// 2x2 average pool on [B, h, w] -> [B, h/2, w/2]
// ---------------------------------------------------------------------------
__global__ void avgpool_kernel(const float* in, float* out, int B, int h, int w) {
  int h2 = h >> 1, w2 = w >> 1;
  int i = blockIdx.x * blockDim.x + threadIdx.x;
  int total = B * h2 * w2;
  if (i >= total) return;
  int x = i % w2;
  int y = (i / w2) % h2;
  int b = i / (w2 * h2);
  const float* p = in + (size_t)b * h * w;
  out[i] = 0.25f * (p[(2 * y) * w + 2 * x]     + p[(2 * y) * w + 2 * x + 1] +
                    p[(2 * y + 1) * w + 2 * x] + p[(2 * y + 1) * w + 2 * x + 1]);
}

// ---------------------------------------------------------------------------
// Correlation pyramid bilinear lookup (align_corners, zero-pad outside)
// out layout [4*81, h, w]; channel = lvl*81 + t, ox = d[t/9], oy = d[t%9]
// ---------------------------------------------------------------------------
__device__ inline float bilin_sample(const float* img, int h, int w, float x, float y) {
  float x0 = floorf(x), y0 = floorf(y);
  float out = 0.f;
#pragma unroll
  for (int dy = 0; dy < 2; ++dy) {
#pragma unroll
    for (int dx = 0; dx < 2; ++dx) {
      float xi = x0 + (float)dx;
      float yi = y0 + (float)dy;
      float wgt = (1.f - fabsf(x - xi)) * (1.f - fabsf(y - yi));
      if ((xi >= 0.f) && (xi <= (float)(w - 1)) && (yi >= 0.f) && (yi <= (float)(h - 1))) {
        int xc = (int)xi, yc = (int)yi;
        out += wgt * img[yc * w + xc];
      }
    }
  }
  return out;
}

__global__ void corr_lookup_kernel(const float* c0, const float* c1,
                                   const float* c2, const float* c3,
                                   const float* coords, float* out, int h, int w)
{
  int idx = blockIdx.x * blockDim.x + threadIdx.x;
  int HW = h * w;
  int total = HW * 4 * 81;
  if (idx >= total) return;
  int p   = idx % HW;
  int t   = (idx / HW) % 81;
  int lvl = idx / (HW * 81);
  float cx = coords[p];
  float cy = coords[HW + p];
  float ox = (float)(t / 9) - 4.f;
  float oy = (float)(t % 9) - 4.f;
  const float* base = (lvl == 0) ? c0 : (lvl == 1) ? c1 : (lvl == 2) ? c2 : c3;
  int hl = h >> lvl, wl = w >> lvl;
  const float* img = base + (size_t)p * hl * wl;
  float scl = 1.f / (float)(1 << lvl);
  out[(size_t)(lvl * 81 + t) * HW + p] =
      bilin_sample(img, hl, wl, cx * scl + ox, cy * scl + oy);
}

// ---------------------------------------------------------------------------
// Convex upsample: mask [576,h,w] (9*8*8), flow [2,h,w] -> out [2,8h,8w]
// ---------------------------------------------------------------------------
__global__ void upsample_kernel(const float* flow, const float* mask,
                                float* out, int h, int w)
{
  int idx = blockIdx.x * blockDim.x + threadIdx.x;
  int HW = h * w;
  int total = HW * 64;
  if (idx >= total) return;
  int p  = idx / 64;
  int ij = idx % 64;
  int i8 = ij / 8, j8 = ij % 8;
  int y = p / w, x = p % w;

  float m[9];
  float mx = -1e30f;
#pragma unroll
  for (int k = 0; k < 9; ++k) {
    m[k] = mask[(size_t)(k * 64 + i8 * 8 + j8) * HW + p];
    mx = fmaxf(mx, m[k]);
  }
  float den = 0.f;
#pragma unroll
  for (int k = 0; k < 9; ++k) { m[k] = expf(m[k] - mx); den += m[k]; }
  float inv = 1.f / den;

  float a0 = 0.f, a1 = 0.f;
#pragma unroll
  for (int k = 0; k < 9; ++k) {
    int di = k / 3 - 1, dj = k % 3 - 1;
    int yy = y + di, xx = x + dj;
    if ((yy >= 0) && (yy < h) && (xx >= 0) && (xx < w)) {
      float wk = m[k] * inv;
      a0 += wk * 8.f * flow[yy * w + xx];
      a1 += wk * 8.f * flow[HW + yy * w + xx];
    }
  }
  int Hup = 8 * h, Wup = 8 * w;
  int Y = y * 8 + i8, X = x * 8 + j8;
  out[(size_t)Y * Wup + X] = a0;
  out[(size_t)Hup * Wup + (size_t)Y * Wup + X] = a1;
}

// ---------------------------------------------------------------------------
// Host-side helpers
// ---------------------------------------------------------------------------
static void launch_conv(hipStream_t s, const float* x, const float* wgt, const float* b,
                        const float* scl, const float* shf, const float* resid, float* y,
                        int C, int H, int W, int O, int KH, int KW, int stride,
                        int relu, float oscale)
{
  int Hout = (H + 2 * (KH / 2) - KH) / stride + 1;
  int Wout = (W + 2 * (KW / 2) - KW) / stride + 1;
  int M = Hout * Wout;
  dim3 grid((M + 15) / 16, (O + 63) / 64);
  conv_wmma_kernel<<<grid, 128, 0, s>>>(x, wgt, b, scl, shf, resid, y,
                                        C, H, W, O, KH, KW, stride,
                                        Hout, Wout, relu, oscale);
}

// Walks the parameter leaves of one encoder (insertion order of the ref dict)
static void run_encoder(hipStream_t s, void* const* d_in, int cur, int bn,
                        const float* img, float* out,
                        float* t0, float* t1, float* t2, float* t3)
{
  const int ci[6] = {64, 64, 64, 96, 96, 128};
  const int co[6] = {64, 64, 96, 96, 128, 128};
  const int st[6] = {1, 1, 2, 1, 2, 1};

  const float* w0 = (const float*)d_in[cur++];
  const float* b0 = (const float*)d_in[cur++];
  const float* n0s = nullptr; const float* n0t = nullptr;
  if (bn) { n0s = (const float*)d_in[cur++]; n0t = (const float*)d_in[cur++]; }

  launch_conv(s, img, w0, b0, n0s, n0t, nullptr, t0,
              3, 384, 768, 64, 7, 7, 2, bn ? 1 : 0, 1.0f);
  if (!bn) instnorm_kernel<<<64, 256, 0, s>>>(t0, t0, 192 * 384, 1, nullptr);

  int H = 192, W = 384;
  float* xb = t0; float* y1 = t1; float* y2 = t2; float* xd = t3;

  for (int bk = 0; bk < 6; ++bk) {
    int C = ci[bk], Co = co[bk], S = st[bk];
    int Ho = (H + 2 - 3) / S + 1;
    int Wo = (W + 2 - 3) / S + 1;
    const float* w1 = (const float*)d_in[cur++];
    const float* b1 = (const float*)d_in[cur++];
    const float* w2 = (const float*)d_in[cur++];
    const float* b2 = (const float*)d_in[cur++];
    const float *n1s = nullptr, *n1t = nullptr, *n2s = nullptr, *n2t = nullptr;
    const float *wd = nullptr, *bd = nullptr, *nds = nullptr, *ndt = nullptr;
    if (bn) {
      n1s = (const float*)d_in[cur++]; n1t = (const float*)d_in[cur++];
      n2s = (const float*)d_in[cur++]; n2t = (const float*)d_in[cur++];
    }
    if (S != 1) {
      wd = (const float*)d_in[cur++]; bd = (const float*)d_in[cur++];
      if (bn) { nds = (const float*)d_in[cur++]; ndt = (const float*)d_in[cur++]; }
    }

    // y1 = relu(norm(conv1(x)))
    launch_conv(s, xb, w1, b1, n1s, n1t, nullptr, y1, C, H, W, Co, 3, 3, S,
                bn ? 1 : 0, 1.0f);
    if (!bn) instnorm_kernel<<<Co, 256, 0, s>>>(y1, y1, Ho * Wo, 1, nullptr);

    // downsample path
    const float* res = xb;
    if (S != 1) {
      launch_conv(s, xb, wd, bd, nds, ndt, nullptr, xd, C, H, W, Co, 1, 1, S, 0, 1.0f);
      if (!bn) instnorm_kernel<<<Co, 256, 0, s>>>(xd, xd, Ho * Wo, 0, nullptr);
      res = xd;
    }

    // x = relu(norm(conv2(y1)) + res)
    if (bn) {
      launch_conv(s, y1, w2, b2, n2s, n2t, res, y2, Co, Ho, Wo, Co, 3, 3, 1, 1, 1.0f);
    } else {
      launch_conv(s, y1, w2, b2, nullptr, nullptr, nullptr, y2, Co, Ho, Wo, Co, 3, 3, 1, 0, 1.0f);
      instnorm_kernel<<<Co, 256, 0, s>>>(y2, y2, Ho * Wo, 1, res);
    }
    float* tmp = xb; xb = y2; y2 = tmp;
    H = Ho; W = Wo;
  }

  const float* wout = (const float*)d_in[cur++];
  const float* bout = (const float*)d_in[cur++];
  launch_conv(s, xb, wout, bout, nullptr, nullptr, nullptr, out,
              128, H, W, 256, 1, 1, 1, 0, 1.0f);
}

static inline dim3 g1(int n) { return dim3((n + 255) / 256); }

// ---------------------------------------------------------------------------
extern "C" void kernel_launch(void* const* d_in, const int* in_sizes, int n_in,
                              void* d_out, int out_size, void* d_ws, size_t ws_size,
                              hipStream_t stream)
{
  (void)in_sizes; (void)n_in; (void)out_size; (void)ws_size;
  const float* image1 = (const float*)d_in[0];
  const float* image2 = (const float*)d_in[1];
  // flat param layout (dict insertion order): fnet=32 leaves, cnet=62, update=30
  const int FNET = 2, CNET = 34, UPD = 96;
  const int iters = 8;

  const int h = 48, w = 96, HW = h * w;

  float* ws = (float*)d_ws;
  size_t off = 0;
  auto alloc = [&](size_t n) {
    float* p = ws + off;
    off += (n + 63) & ~(size_t)63;
    return p;
  };

  const size_t ACT = (size_t)64 * 192 * 384;
  float* t0 = alloc(ACT);
  float* t1 = alloc(ACT);
  float* t2 = alloc(ACT);
  float* t3 = alloc(ACT);
  float* fmap1 = alloc((size_t)256 * HW);
  float* fmap2 = alloc((size_t)256 * HW);
  float* cnetb = alloc((size_t)256 * HW);
  float* pyr0 = alloc((size_t)HW * HW);
  float* pyr1 = alloc((size_t)HW * (HW / 4));
  float* pyr2 = alloc((size_t)HW * (HW / 16));
  float* pyr3 = alloc((size_t)HW * (HW / 64));
  float* coords1 = alloc((size_t)2 * HW);
  float* netb = alloc((size_t)128 * HW);
  float* gru  = alloc((size_t)384 * HW);   // [slot(128) | inp(128) | mf(126) | flow(2)]
  float* Xinp  = gru + (size_t)128 * HW;
  float* Xmf   = gru + (size_t)256 * HW;
  float* Xflow = gru + (size_t)382 * HW;
  float* corrlk = alloc((size_t)324 * HW);
  float* cor1 = alloc((size_t)256 * HW);
  float* CF   = alloc((size_t)256 * HW);   // [cor(192) | flo(64)]
  float* flo1 = alloc((size_t)128 * HW);
  float* zc = alloc((size_t)128 * HW);
  float* rc = alloc((size_t)128 * HW);
  float* qc = alloc((size_t)128 * HW);
  float* htmp  = alloc((size_t)256 * HW);
  float* delta = alloc((size_t)2 * HW);
  float* maskb = alloc((size_t)576 * HW);

  // ---- encoders ----
  run_encoder(stream, d_in, FNET, 0, image1, fmap1, t0, t1, t2, t3);
  run_encoder(stream, d_in, FNET, 0, image2, fmap2, t0, t1, t2, t3);
  run_encoder(stream, d_in, CNET, 1, image1, cnetb, t0, t1, t2, t3);

  // ---- correlation pyramid ----
  {
    dim3 grid((HW + 15) / 16, (HW + 63) / 64);
    gemm_tn_wmma_kernel<<<grid, 128, 0, stream>>>(fmap1, fmap2, pyr0,
                                                  HW, HW, 256, 1.0f / 16.0f);
    avgpool_kernel<<<g1(HW * (HW / 4)),  256, 0, stream>>>(pyr0, pyr1, HW, 48, 96);
    avgpool_kernel<<<g1(HW * (HW / 16)), 256, 0, stream>>>(pyr1, pyr2, HW, 24, 48);
    avgpool_kernel<<<g1(HW * (HW / 64)), 256, 0, stream>>>(pyr2, pyr3, HW, 12, 24);
  }

  // ---- context split: net = tanh(cnet[:128]), inp = relu(cnet[128:]) ----
  ew_tanh_kernel<<<g1(128 * HW), 256, 0, stream>>>(netb, cnetb, 128 * HW);
  ew_relu_kernel<<<g1(128 * HW), 256, 0, stream>>>(Xinp, cnetb + (size_t)128 * HW, 128 * HW);
  coords_init_kernel<<<g1(HW), 256, 0, stream>>>(coords1, h, w);

  // ---- update-block params ----
  int cu = UPD;
  auto nx = [&]() { return (const float*)d_in[cu++]; };
  const float *wc1 = nx(), *bc1 = nx(), *wc2 = nx(), *bc2 = nx();
  const float *wf1 = nx(), *bf1 = nx(), *wf2 = nx(), *bf2 = nx();
  const float *wcf = nx(), *bcf = nx();
  const float *wz1 = nx(), *bz1 = nx(), *wr1 = nx(), *br1 = nx(), *wq1 = nx(), *bq1 = nx();
  const float *wz2 = nx(), *bz2 = nx(), *wr2 = nx(), *br2 = nx(), *wq2 = nx(), *bq2 = nx();
  const float *wfh1 = nx(), *bfh1 = nx(), *wfh2 = nx(), *bfh2 = nx();
  const float *wm1 = nx(), *bm1 = nx(), *wm2 = nx(), *bm2 = nx();

  for (int it = 0; it < iters; ++it) {
    corr_lookup_kernel<<<g1(HW * 324), 256, 0, stream>>>(pyr0, pyr1, pyr2, pyr3,
                                                         coords1, corrlk, h, w);
    flow_from_coords_kernel<<<g1(HW), 256, 0, stream>>>(coords1, Xflow, h, w);

    // motion encoder
    launch_conv(stream, corrlk, wc1, bc1, nullptr, nullptr, nullptr, cor1,
                324, h, w, 256, 1, 1, 1, 1, 1.0f);
    launch_conv(stream, cor1, wc2, bc2, nullptr, nullptr, nullptr, CF,
                256, h, w, 192, 3, 3, 1, 1, 1.0f);
    launch_conv(stream, Xflow, wf1, bf1, nullptr, nullptr, nullptr, flo1,
                2, h, w, 128, 7, 7, 1, 1, 1.0f);
    launch_conv(stream, flo1, wf2, bf2, nullptr, nullptr, nullptr,
                CF + (size_t)192 * HW, 128, h, w, 64, 3, 3, 1, 1, 1.0f);
    launch_conv(stream, CF, wcf, bcf, nullptr, nullptr, nullptr, Xmf,
                256, h, w, 126, 3, 3, 1, 1, 1.0f);

    // SepConvGRU pass 1 (1x5)
    ew_copy_kernel<<<g1(128 * HW), 256, 0, stream>>>(gru, netb, 128 * HW);
    launch_conv(stream, gru, wz1, bz1, nullptr, nullptr, nullptr, zc,
                384, h, w, 128, 1, 5, 1, 0, 1.0f);
    launch_conv(stream, gru, wr1, br1, nullptr, nullptr, nullptr, rc,
                384, h, w, 128, 1, 5, 1, 0, 1.0f);
    ew_sigmul_kernel<<<g1(128 * HW), 256, 0, stream>>>(gru, rc, netb, 128 * HW);
    launch_conv(stream, gru, wq1, bq1, nullptr, nullptr, nullptr, qc,
                384, h, w, 128, 1, 5, 1, 0, 1.0f);
    ew_gru_kernel<<<g1(128 * HW), 256, 0, stream>>>(netb, zc, qc, 128 * HW);

    // SepConvGRU pass 2 (5x1)
    ew_copy_kernel<<<g1(128 * HW), 256, 0, stream>>>(gru, netb, 128 * HW);
    launch_conv(stream, gru, wz2, bz2, nullptr, nullptr, nullptr, zc,
                384, h, w, 128, 5, 1, 1, 0, 1.0f);
    launch_conv(stream, gru, wr2, br2, nullptr, nullptr, nullptr, rc,
                384, h, w, 128, 5, 1, 1, 0, 1.0f);
    ew_sigmul_kernel<<<g1(128 * HW), 256, 0, stream>>>(gru, rc, netb, 128 * HW);
    launch_conv(stream, gru, wq2, bq2, nullptr, nullptr, nullptr, qc,
                384, h, w, 128, 5, 1, 1, 0, 1.0f);
    ew_gru_kernel<<<g1(128 * HW), 256, 0, stream>>>(netb, zc, qc, 128 * HW);

    // flow head
    launch_conv(stream, netb, wfh1, bfh1, nullptr, nullptr, nullptr, htmp,
                128, h, w, 256, 3, 3, 1, 1, 1.0f);
    launch_conv(stream, htmp, wfh2, bfh2, nullptr, nullptr, nullptr, delta,
                256, h, w, 2, 3, 3, 1, 0, 1.0f);
    // mask head (0.25 folded into epilogue)
    launch_conv(stream, netb, wm1, bm1, nullptr, nullptr, nullptr, htmp,
                128, h, w, 256, 3, 3, 1, 1, 1.0f);
    launch_conv(stream, htmp, wm2, bm2, nullptr, nullptr, nullptr, maskb,
                256, h, w, 576, 1, 1, 1, 0, 0.25f);

    ew_add_kernel<<<g1(2 * HW), 256, 0, stream>>>(coords1, delta, 2 * HW);
  }

  // outputs: [flow_lr (2*48*96) | flow_up (2*384*768)]
  float* flow_lr = (float*)d_out;
  float* flow_up = flow_lr + (size_t)2 * HW;
  flow_from_coords_kernel<<<g1(HW), 256, 0, stream>>>(coords1, flow_lr, h, w);
  upsample_kernel<<<g1(HW * 64), 256, 0, stream>>>(flow_lr, maskb, flow_up, h, w);
}
